// MatchWindowAttention_214748365223
// MI455X (gfx1250) — matrix operand order repaired and checked by reading, not compiled, not run on hardware
//
#include <hip/hip_runtime.h>
#include <math.h>

typedef __bf16 bf16;
typedef __attribute__((ext_vector_type(16))) __bf16 v16bf;
typedef __attribute__((ext_vector_type(8)))  __bf16 v8bf;
typedef __attribute__((ext_vector_type(4)))  __bf16 v4bf;
typedef __attribute__((ext_vector_type(8)))  float  v8f;
typedef __attribute__((ext_vector_type(4)))  float  v4f;

#define NTOK   65536        // B * H * W
#define DMODEL 256
#define QKVLD  768
#define FFNH   512
#define NWIN   1024
#define NHEAD  4
#define DH     64

__device__ __forceinline__ v8bf ld8(const bf16* p) {
    return *reinterpret_cast<const v8bf*>(p);
}
__device__ __forceinline__ v4f ld4f(const float* p) {
    return *reinterpret_cast<const v4f*>(p);
}

// A-fragment for V_WMMA_*_16x16x32 (16-bit): lane<16 holds K {0..7,16..23},
// lane>=16 holds K {8..15,24..31} of row (lane&15). `row` points at A[m][k0].
__device__ __forceinline__ v16bf make_afrag(const bf16* row, int half) {
    v8bf lo = ld8(row + half * 8);
    v8bf hi = ld8(row + 16 + half * 8);
    v16bf a;
#pragma unroll
    for (int i = 0; i < 8; ++i) { a[i] = lo[i]; a[i + 8] = hi[i]; }
    return a;
}

// B-fragment (32x16, K-major): lane L holds column (L&15), K {0..7,16..23} for
// half 0, K {8..15,24..31} for half 1.  `col` points at B-column start = W-row start + k0.
__device__ __forceinline__ v16bf make_bfrag(const bf16* col, int half) {
    v8bf lo = ld8(col + half * 8);
    v8bf hi = ld8(col + 16 + half * 8);
    v16bf b;
#pragma unroll
    for (int i = 0; i < 8; ++i) { b[i] = lo[i]; b[i + 8] = hi[i]; }
    return b;
}

__device__ __forceinline__ v8f wmma_bf16(v16bf a, v16bf b, v8f c) {
    return __builtin_amdgcn_wmma_f32_16x16x32_bf16(
        false, a, false, b, (short)0, c, false, false);
}

// ---------------------------------------------------------------- weights f32 -> bf16
// 4 elements per thread, b128 load / b64 store.
__global__ void f2bf_kernel(const float* __restrict__ in, bf16* __restrict__ out, int n4) {
    int i = blockIdx.x * 256 + threadIdx.x;
    if (i < n4) {
        v4f v = ld4f(in + (size_t)i * 4);
        v4bf o;
#pragma unroll
        for (int j = 0; j < 4; ++j) o[j] = (bf16)v[j];
        *reinterpret_cast<v4bf*>(out + (size_t)i * 4) = o;
    }
}

// ------------------------------------------------- window partition + positional enc
// feat [16, 4096, 256] f32 (image order) -> win [65536, 256] bf16 (window order)
// 4 channels per thread.
__global__ void partition_pe_kernel(const float* __restrict__ feat,
                                    const float* __restrict__ pe,
                                    bf16* __restrict__ win) {
    size_t idx = (size_t)blockIdx.x * 256 + threadIdx.x;   // over 65536*64
    int c4 = (int)(idx & 63);                              // group of 4 channels
    size_t row = idx >> 6;
    int tw = (int)(row & 63);           // token within window
    size_t w = row >> 6;                // window id = ((b*8+wy)*8+wx)
    int wx = (int)(w & 7);
    int wy = (int)((w >> 3) & 7);
    int b  = (int)(w >> 6);
    int iy = tw >> 3, ix = tw & 7;
    int p  = (wy * 8 + iy) * 64 + (wx * 8 + ix);
    v4f f = ld4f(feat + (((size_t)b * 4096 + p) * DMODEL + c4 * 4));
    v4f q = ld4f(pe + ((size_t)tw * DMODEL + c4 * 4));
    v4bf o;
#pragma unroll
    for (int j = 0; j < 4; ++j) o[j] = (bf16)(f[j] + q[j]);
    *reinterpret_cast<v4bf*>(win + idx * 4) = o;
}

// ------------------------------------------------------------------ generic WMMA GEMM
// C[M,N] = A[M,K](bf16,lda) @ W[N,K]^T (bf16) + bias, epilogue: optional GELU,
// output bf16 or f32 with leading dim ldc.  Block: 256 thr (8 waves), each wave
// computes a 16x64 strip.  Grid: (M/128, N/64).  K is compile-time: the fully
// unrolled k-loop lets the scheduler hoist next-step b128 loads over WMMAs.
template <int K, bool OUT_BF16, bool GELU>
__global__ void gemm_wmma_kernel(const bf16* __restrict__ A, int lda,
                                 const bf16* __restrict__ W,
                                 const float* __restrict__ bias,
                                 void* __restrict__ Cout, int ldc) {
    const int wave = threadIdx.x >> 5;
    const int lane = threadIdx.x & 31;
    const int half = lane >> 4;
    const int r    = lane & 15;
    const int m0 = blockIdx.x * 128 + wave * 16;
    const int n0 = blockIdx.y * 64;

    v8f acc[4];
#pragma unroll
    for (int t = 0; t < 4; ++t)
#pragma unroll
        for (int j = 0; j < 8; ++j) acc[t][j] = 0.f;

    const bf16* arow = A + (size_t)(m0 + r) * lda;
    const bf16* wbase = W + (size_t)(n0 + r) * K;
#pragma unroll
    for (int k0 = 0; k0 < K; k0 += 32) {
        v16bf af = make_afrag(arow + k0, half);
#pragma unroll
        for (int t = 0; t < 4; ++t) {
            v16bf bf_ = make_bfrag(wbase + (size_t)(16 * t) * K + k0, half);
            acc[t] = wmma_bf16(af, bf_, acc[t]);
        }
    }

#pragma unroll
    for (int t = 0; t < 4; ++t) {
        int n = n0 + 16 * t + r;
        float bv = bias[n];
#pragma unroll
        for (int j = 0; j < 8; ++j) {
            int m = m0 + j + 8 * half;
            float v = acc[t][j] + bv;
            if (GELU) v = 0.5f * v * (1.f + erff(v * 0.70710678118654752f));
            if (OUT_BF16) ((bf16*)Cout)[(size_t)m * ldc + n] = (bf16)v;
            else          ((float*)Cout)[(size_t)m * ldc + n] = v;
        }
    }
}

// ------------------------------------------------------------- window cross-attention
// One block per (window, head).  q from qkv_q, k/v from qkv_kv (same window).
// qkv rows: [q(0..255) | k(256..511) | v(512..767)], head h at +h*64.
__global__ void attn_kernel(const bf16* __restrict__ qkv_q,
                            const bf16* __restrict__ qkv_kv,
                            bf16* __restrict__ out) {
    const int win  = blockIdx.x;
    const int head = blockIdx.y;
    const int wave = threadIdx.x >> 5;
    const int lane = threadIdx.x & 31;
    const int half = lane >> 4;
    const int r    = lane & 15;

    __shared__ bf16 vT[64][72];      // v transposed: vT[d][k]
    __shared__ bf16 P[4][16][72];    // per-wave softmax probs strip

    const bf16* qbase = qkv_q  + (size_t)win * 64 * QKVLD + head * DH;
    const bf16* kbase = qkv_kv + (size_t)win * 64 * QKVLD + 256 + head * DH;
    const bf16* vbase = qkv_kv + (size_t)win * 64 * QKVLD + 512 + head * DH;

    for (int i = threadIdx.x; i < 64 * 64; i += 128) {
        int k = i >> 6, d = i & 63;
        vT[d][k] = vbase[(size_t)k * QKVLD + d];
    }
    __syncthreads();

    const int m0 = wave * 16;

    // S = Q K^T  (16x64 strip per wave)
    v8f s[4];
#pragma unroll
    for (int t = 0; t < 4; ++t)
#pragma unroll
        for (int j = 0; j < 8; ++j) s[t][j] = 0.f;

#pragma unroll
    for (int k0 = 0; k0 < DH; k0 += 32) {
        v16bf af = make_afrag(qbase + (size_t)(m0 + r) * QKVLD + k0, half);
#pragma unroll
        for (int t = 0; t < 4; ++t) {
            v16bf bf_ = make_bfrag(kbase + (size_t)(16 * t + r) * QKVLD + k0, half);
            s[t] = wmma_bf16(af, bf_, s[t]);
        }
    }

    // row softmax in C-layout: row = j + 8*half, spread over 16 lanes x 4 tiles
    const float scale = 0.125f; // 1/sqrt(64)
#pragma unroll
    for (int j = 0; j < 8; ++j) {
        float m = fmaxf(fmaxf(s[0][j], s[1][j]), fmaxf(s[2][j], s[3][j]));
#pragma unroll
        for (int off = 1; off <= 8; off <<= 1) m = fmaxf(m, __shfl_xor(m, off, 32));
        float e[4], sum = 0.f;
#pragma unroll
        for (int t = 0; t < 4; ++t) { e[t] = __expf((s[t][j] - m) * scale); sum += e[t]; }
#pragma unroll
        for (int off = 1; off <= 8; off <<= 1) sum += __shfl_xor(sum, off, 32);
        float inv = 1.f / sum;
#pragma unroll
        for (int t = 0; t < 4; ++t)
            P[wave][j + 8 * half][16 * t + r] = (bf16)(e[t] * inv);
    }
    __syncthreads();

    // O = P V  (16x64 strip)
    v8f o[4];
#pragma unroll
    for (int t = 0; t < 4; ++t)
#pragma unroll
        for (int j = 0; j < 8; ++j) o[t][j] = 0.f;

#pragma unroll
    for (int k0 = 0; k0 < 64; k0 += 32) {
        v16bf af = make_afrag(&P[wave][r][k0], half);
#pragma unroll
        for (int t = 0; t < 4; ++t) {
            v16bf bf_ = make_bfrag(&vT[16 * t + r][k0], half);
            o[t] = wmma_bf16(af, bf_, o[t]);
        }
    }

    bf16* obase = out + (size_t)win * 64 * DMODEL + head * DH;
#pragma unroll
    for (int t = 0; t < 4; ++t)
#pragma unroll
        for (int j = 0; j < 8; ++j)
            obase[(size_t)(m0 + j + 8 * half) * DMODEL + 16 * t + r] = (bf16)o[t][j];
}

// ------------------------------------------------- residual + LayerNorm (1 wave/token)
// GATHER: x1 is in window order (out-proj result), gathered via window-reverse.
// Vectorized: 2x float4 per lane per source (lane*4 and 128+lane*4).
template <bool GATHER, bool EMIT_BF16>
__global__ void ln_kernel(const float* __restrict__ x1,
                          const float* __restrict__ x2,
                          const float* __restrict__ g,
                          const float* __restrict__ beta,
                          float* __restrict__ out_f32,
                          bf16* __restrict__ out_bf16) {
    int token = blockIdx.x * 8 + (threadIdx.x >> 5);
    int lane  = threadIdx.x & 31;

    const float* s1;
    if (GATHER) {
        int b  = token >> 12;
        int p  = token & 4095;
        int yy = p >> 6, xx = p & 63;
        int wr = ((b * 8 + (yy >> 3)) * 8 + (xx >> 3)) * 64 + ((yy & 7) * 8 + (xx & 7));
        s1 = x1 + (size_t)wr * DMODEL;
    } else {
        s1 = x1 + (size_t)token * DMODEL;
    }
    const float* s2 = x2 + (size_t)token * DMODEL;

    int c0 = lane * 4;          // first quad
    int c1 = 128 + lane * 4;    // second quad
    v4f a0 = ld4f(s1 + c0), a1 = ld4f(s1 + c1);
    v4f b0 = ld4f(s2 + c0), b1 = ld4f(s2 + c1);
    float vals[8];
    float sum = 0.f;
#pragma unroll
    for (int i = 0; i < 4; ++i) {
        vals[i]     = a0[i] + b0[i];
        vals[i + 4] = a1[i] + b1[i];
        sum += vals[i] + vals[i + 4];
    }
#pragma unroll
    for (int off = 1; off <= 16; off <<= 1) sum += __shfl_xor(sum, off, 32);
    float mean = sum * (1.f / 256.f);
    float var = 0.f;
#pragma unroll
    for (int i = 0; i < 8; ++i) { float d = vals[i] - mean; var += d * d; }
#pragma unroll
    for (int off = 1; off <= 16; off <<= 1) var += __shfl_xor(var, off, 32);
    float rs = rsqrtf(var * (1.f / 256.f) + 1e-5f);

    v4f g0 = ld4f(g + c0), g1 = ld4f(g + c1);
    v4f t0 = ld4f(beta + c0), t1 = ld4f(beta + c1);
    v4f o0, o1;
#pragma unroll
    for (int i = 0; i < 4; ++i) {
        o0[i] = (vals[i]     - mean) * rs * g0[i] + t0[i];
        o1[i] = (vals[i + 4] - mean) * rs * g1[i] + t1[i];
    }
    float* dst = out_f32 + (size_t)token * DMODEL;
    *reinterpret_cast<v4f*>(dst + c0) = o0;
    *reinterpret_cast<v4f*>(dst + c1) = o1;
    if (EMIT_BF16) {
        v4bf p0, p1;
#pragma unroll
        for (int i = 0; i < 4; ++i) { p0[i] = (bf16)o0[i]; p1[i] = (bf16)o1[i]; }
        bf16* bdst = out_bf16 + (size_t)token * DMODEL;
        *reinterpret_cast<v4bf*>(bdst + c0) = p0;
        *reinterpret_cast<v4bf*>(bdst + c1) = p1;
    }
}

// =====================================================================================
extern "C" void kernel_launch(void* const* d_in, const int* in_sizes, int n_in,
                              void* d_out, int out_size, void* d_ws, size_t ws_size,
                              hipStream_t stream) {
    (void)in_sizes; (void)n_in; (void)out_size; (void)ws_size;

    const float* feat[2] = { (const float*)d_in[0], (const float*)d_in[1] };
    const float* pe = (const float*)d_in[2];
    // per-stream params start at 3 (A) and 15 (B)
    const float *qkv_w[2], *qkv_b[2], *out_w[2], *out_b[2];
    const float *norm_g[2], *norm_b[2], *fnorm_g[2], *fnorm_b[2];
    const float *w1[2], *b1[2], *w2[2], *b2[2];
    for (int s = 0; s < 2; ++s) {
        int base = 3 + s * 12;
        qkv_w[s]  = (const float*)d_in[base + 0];
        qkv_b[s]  = (const float*)d_in[base + 1];
        out_w[s]  = (const float*)d_in[base + 2];
        out_b[s]  = (const float*)d_in[base + 3];
        norm_g[s] = (const float*)d_in[base + 4];
        norm_b[s] = (const float*)d_in[base + 5];
        fnorm_g[s]= (const float*)d_in[base + 6];
        fnorm_b[s]= (const float*)d_in[base + 7];
        w1[s]     = (const float*)d_in[base + 8];
        b1[s]     = (const float*)d_in[base + 9];
        w2[s]     = (const float*)d_in[base + 10];
        b2[s]     = (const float*)d_in[base + 11];
    }

    // ---- workspace layout (lifetime-aliased) ----
    char* ws = (char*)d_ws;
    size_t off = 0;
    auto take = [&](size_t bytes) { size_t o = off; off += (bytes + 255) & ~(size_t)255; return o; };

    size_t o_wq[2], o_wo[2], o_w1[2], o_w2[2];
    for (int s = 0; s < 2; ++s) {
        o_wq[s] = take((size_t)QKVLD * DMODEL * 2);
        o_wo[s] = take((size_t)DMODEL * DMODEL * 2);
        o_w1[s] = take((size_t)FFNH  * DMODEL * 2);
        o_w2[s] = take((size_t)DMODEL * FFNH  * 2);
    }
    size_t o_win = take((size_t)2 * NTOK * DMODEL * 2);   // win -> attn -> ln1_bf16
    size_t o_big = take((size_t)2 * NTOK * QKVLD * 2);    // qkv -> y(f32) -> h(bf16)
    size_t o_ln1 = take((size_t)2 * NTOK * DMODEL * 4);   // ln1 f32
    size_t o_x2  = take((size_t)2 * NTOK * DMODEL * 4);   // ffn2 out f32

    bf16* win[2]  = { (bf16*)(ws + o_win), (bf16*)(ws + o_win) + (size_t)NTOK * DMODEL };
    bf16* attn[2] = { win[0], win[1] };
    bf16* ln1b[2] = { win[0], win[1] };
    bf16* qkv[2]  = { (bf16*)(ws + o_big), (bf16*)(ws + o_big) + (size_t)NTOK * QKVLD };
    float* y[2]   = { (float*)(ws + o_big), (float*)(ws + o_big) + (size_t)NTOK * DMODEL };
    bf16* h[2]    = { (bf16*)(ws + o_big), (bf16*)(ws + o_big) + (size_t)NTOK * FFNH };
    float* ln1[2] = { (float*)(ws + o_ln1), (float*)(ws + o_ln1) + (size_t)NTOK * DMODEL };
    float* x2[2]  = { (float*)(ws + o_x2),  (float*)(ws + o_x2)  + (size_t)NTOK * DMODEL };
    float* outp[2]= { (float*)d_out, (float*)d_out + (size_t)NTOK * DMODEL };

    auto cvt = [&](const float* src, size_t o, int n) {
        int n4 = n / 4;
        f2bf_kernel<<<(n4 + 255) / 256, 256, 0, stream>>>(src, (bf16*)(ws + o), n4);
    };

    // 1) weights -> bf16
    for (int s = 0; s < 2; ++s) {
        cvt(qkv_w[s], o_wq[s], QKVLD * DMODEL);
        cvt(out_w[s], o_wo[s], DMODEL * DMODEL);
        cvt(w1[s],    o_w1[s], FFNH * DMODEL);
        cvt(w2[s],    o_w2[s], DMODEL * FFNH);
    }

    // 2) partition + PE (4 channels / thread)
    for (int s = 0; s < 2; ++s)
        partition_pe_kernel<<<NTOK / 4, 256, 0, stream>>>(feat[s], pe, win[s]);

    // 3) fused QKV GEMM: [65536,256] @ [768,256]^T -> bf16 [65536,768]
    for (int s = 0; s < 2; ++s)
        gemm_wmma_kernel<DMODEL, true, false><<<dim3(NTOK / 128, QKVLD / 64), 256, 0, stream>>>(
            win[s], DMODEL, (const bf16*)(ws + o_wq[s]), qkv_b[s], qkv[s], QKVLD);

    // 4) cross attention per (window, head): A queries B's K/V and vice versa
    attn_kernel<<<dim3(NWIN, NHEAD), 128, 0, stream>>>(qkv[0], qkv[1], attn[0]);
    attn_kernel<<<dim3(NWIN, NHEAD), 128, 0, stream>>>(qkv[1], qkv[0], attn[1]);

    // 5) out-proj GEMM -> f32 (window order)
    for (int s = 0; s < 2; ++s)
        gemm_wmma_kernel<DMODEL, false, false><<<dim3(NTOK / 128, DMODEL / 64), 256, 0, stream>>>(
            attn[s], DMODEL, (const bf16*)(ws + o_wo[s]), out_b[s], y[s], DMODEL);

    // 6) window-reverse + residual + LN1 -> f32 + bf16 (image order)
    for (int s = 0; s < 2; ++s)
        ln_kernel<true, true><<<NTOK / 8, 256, 0, stream>>>(
            y[s], feat[s], norm_g[s], norm_b[s], ln1[s], ln1b[s]);

    // 7) FFN1 GEMM + exact GELU -> bf16 [65536,512]
    for (int s = 0; s < 2; ++s)
        gemm_wmma_kernel<DMODEL, true, true><<<dim3(NTOK / 128, FFNH / 64), 256, 0, stream>>>(
            ln1b[s], DMODEL, (const bf16*)(ws + o_w1[s]), b1[s], h[s], FFNH);

    // 8) FFN2 GEMM -> f32 [65536,256]
    for (int s = 0; s < 2; ++s)
        gemm_wmma_kernel<FFNH, false, false><<<dim3(NTOK / 128, DMODEL / 64), 256, 0, stream>>>(
            h[s], FFNH, (const bf16*)(ws + o_w2[s]), b2[s], x2[s], DMODEL);

    // 9) residual + LN2 -> d_out
    for (int s = 0; s < 2; ++s)
        ln_kernel<false, false><<<NTOK / 8, 256, 0, stream>>>(
            ln1[s], x2[s], fnorm_g[s], fnorm_b[s], outp[s], nullptr);
}
